// DeltaNet_31877247271474
// MI455X (gfx1250) — matrix operand order repaired and checked by reading, not compiled, not run on hardware
//
#include <hip/hip_runtime.h>
#include <math.h>

// ---------------------------------------------------------------- types
typedef __attribute__((ext_vector_type(16))) _Float16 v16h;
typedef __attribute__((ext_vector_type(8)))  _Float16 v8h;
typedef __attribute__((ext_vector_type(4)))  _Float16 v4h;
typedef __attribute__((ext_vector_type(8)))  float    v8f;
typedef __attribute__((ext_vector_type(4)))  float    v4f;

#define B_   2
#define L_   4096
#define H_   1024
#define NH_  4
#define DK_  256
#define DV_  256
#define CH_  32                 // CHUNK
#define NCH_ (L_ / CH_)         // 128 chunks
#define DVB  64                 // dv slice per delta workgroup
#define NDVB (DV_ / DVB)

// ---------------------------------------------------------------- WMMA helpers
// wave32 fragment layouts used here (contiguous k order, same for A and B). LDS layouts are chosen so each
// lane's fragment elements are CONTIGUOUS -> ds_load_b128:
//   A operand: M-major source; lane m=lane%16, run  [k0+16*hh,+16).
//   B operand: N-major source; lane n=lane%16, run  [k0+16*hh,+16).
//   C/D      : N-major f32;    lane n=lane%16, rows [m0+8*hh,+8).

__device__ __forceinline__ v8f wmma_f16(v16h a, v16h b, v8f c) {
  return __builtin_amdgcn_wmma_f32_16x16x32_f16(false, a, false, b, (short)0, c,
                                                false, false);
}

__device__ __forceinline__ v16h pack16(v8h lo, v8h hi) {
  v16h r;
#pragma unroll
  for (int j = 0; j < 8; ++j) { r[j] = lo[j]; r[8 + j] = hi[j]; }
  return r;
}

__device__ __forceinline__ v16h ldA(const _Float16* p, int ld, int m0, int k0,
                                    int lane) {
  const _Float16* row = p + (m0 + (lane & 15)) * ld + k0 + 16 * (lane >> 4);
  return pack16(*(const v8h*)row, *(const v8h*)(row + 8));
}

__device__ __forceinline__ v16h ldA_rs(const _Float16* p, int ld, int m0, int k0,
                                       const float* sc, int lane) {
  int m = m0 + (lane & 15);
  const _Float16* row = p + m * ld + k0 + 16 * (lane >> 4);
  v16h a = pack16(*(const v8h*)row, *(const v8h*)(row + 8));
  _Float16 s = (_Float16)sc[m];
#pragma unroll
  for (int j = 0; j < 16; ++j) a[j] *= s;
  return a;
}

__device__ __forceinline__ v16h ldB(const _Float16* p, int ld, int n0, int k0,
                                    int lane) {
  const _Float16* row = p + (n0 + (lane & 15)) * ld + k0 + 16 * (lane >> 4);
  return pack16(*(const v8h*)row, *(const v8h*)(row + 8));
}

__device__ __forceinline__ v16h ldB_ks(const _Float16* p, int ld, int n0, int k0,
                                       const float* sc, int lane) {
  int kb = k0 + 16 * (lane >> 4);
  const _Float16* row = p + (n0 + (lane & 15)) * ld + kb;
  v16h b = pack16(*(const v8h*)row, *(const v8h*)(row + 8));
#pragma unroll
  for (int j = 0; j < 16; ++j) b[j] *= (_Float16)sc[kb + j];
  return b;
}

__device__ __forceinline__ v16h ldB_f32(const float* p, int ld, int n0, int k0,
                                        int lane) {
  const float* row = p + (n0 + (lane & 15)) * ld + k0 + 16 * (lane >> 4);
  v4f x0 = *(const v4f*)row, x1 = *(const v4f*)(row + 4);
  v4f x2 = *(const v4f*)(row + 8), x3 = *(const v4f*)(row + 12);
  v16h b;
#pragma unroll
  for (int j = 0; j < 4; ++j) {
    b[j] = (_Float16)x0[j];
    b[4 + j] = (_Float16)x1[j];
    b[8 + j] = (_Float16)x2[j];
    b[12 + j] = (_Float16)x3[j];
  }
  return b;
}

__device__ __forceinline__ v8f ldC_nm(const float* p, int ld, int m0, int n0,
                                      int lane) {
  const float* q = p + (n0 + (lane & 15)) * ld + m0 + 8 * (lane >> 4);
  v4f a = *(const v4f*)q, b = *(const v4f*)(q + 4);
  v8f c;
#pragma unroll
  for (int r = 0; r < 4; ++r) { c[r] = a[r]; c[4 + r] = b[r]; }
  return c;
}

__device__ __forceinline__ void stC_nm(float* p, int ld, int m0, int n0, int lane,
                                       v8f c) {
  float* q = p + (n0 + (lane & 15)) * ld + m0 + 8 * (lane >> 4);
  v4f a, b;
#pragma unroll
  for (int r = 0; r < 4; ++r) { a[r] = c[r]; b[r] = c[4 + r]; }
  *(v4f*)q = a;
  *(v4f*)(q + 4) = b;
}

__device__ __forceinline__ float wred(float x) {
#pragma unroll
  for (int o = 16; o > 0; o >>= 1) x += __shfl_xor(x, o, 32);
  return x;
}

__device__ __forceinline__ float sigm(float x) { return 1.f / (1.f + expf(-x)); }

// ---------------------------------------------------------------- weight pre-transpose
// Wt[n*K + k] = (f16)W[k*N + n]  (LDS-tiled so both sides are coalesced).
__global__ __launch_bounds__(256) void wtrans_kernel(const float* __restrict__ W,
                                                     _Float16* __restrict__ Wt,
                                                     int K, int N) {
  __shared__ float tile[32][33];
  int kb = blockIdx.x * 32, nb = blockIdx.y * 32;
  int tx = threadIdx.x & 31, ty = threadIdx.x >> 5;
  for (int i = ty; i < 32; i += 8)
    tile[i][tx] = W[(size_t)(kb + i) * N + nb + tx];
  __syncthreads();
  for (int i = ty; i < 32; i += 8)
    Wt[(size_t)(nb + i) * K + kb + tx] = (_Float16)tile[tx][i];
}

// ---------------------------------------------------------------- pipelined WMMA GEMM
// C[M,N] = A[M,K] * Bt^T where Bt is f16 N-major [N][K] (pre-transposed weight).
// Block tile 128x256, 8 waves of 64x64 (4x4 fragments). Register-staged double
// buffering: global loads for tile k+1 issue while WMMAs consume tile k.
// Requires M%128==0, N%256==0, K%32==0.
__global__ __launch_bounds__(256) void gemm_wmma_kernel(
    const float* __restrict__ A, const _Float16* __restrict__ Bt,
    float* __restrict__ C, int M, int N, int K) {
  __shared__ alignas(16) _Float16 sA[128 * 32];  // M-major
  __shared__ alignas(16) _Float16 sB[256 * 32];  // N-major
  const int tid = threadIdx.x, lane = tid & 31, wid = tid >> 5;
  const int m0 = blockIdx.y * 128, n0 = blockIdx.x * 256;
  const int wm = (wid & 1) * 64, wn = (wid >> 1) * 64;

  int ar[4], ac[4], bn[4], bk[4];
#pragma unroll
  for (int i = 0; i < 4; ++i) {
    int idx = tid + 256 * i;
    ar[i] = idx >> 3;       // A: 128 rows x 8 v4f chunks
    ac[i] = (idx & 7) * 4;
    bn[i] = idx >> 2;       // B: 256 rows x 4 v8h chunks
    bk[i] = (idx & 3) * 8;
  }
  v4f ra[4];
  v8h rbv[4];

  auto fetch = [&](int kk) {
#pragma unroll
    for (int i = 0; i < 4; ++i)
      ra[i] = *(const v4f*)(A + (size_t)(m0 + ar[i]) * K + kk + ac[i]);
#pragma unroll
    for (int i = 0; i < 4; ++i)
      rbv[i] = *(const v8h*)(Bt + (size_t)(n0 + bn[i]) * K + kk + bk[i]);
  };
  auto stage = [&]() {
#pragma unroll
    for (int i = 0; i < 4; ++i) {
      v4h hx;
#pragma unroll
      for (int j = 0; j < 4; ++j) hx[j] = (_Float16)ra[i][j];
      *(v4h*)(sA + ar[i] * 32 + ac[i]) = hx;
    }
#pragma unroll
    for (int i = 0; i < 4; ++i) *(v8h*)(sB + bn[i] * 32 + bk[i]) = rbv[i];
  };

  v8f acc[4][4] = {};
  fetch(0);
  stage();
  __syncthreads();
  for (int kk = 0; kk < K; kk += 32) {
    bool more = (kk + 32) < K;
    if (more) fetch(kk + 32);  // in flight during the WMMAs below
    v16h bf[4];
#pragma unroll
    for (int fn = 0; fn < 4; ++fn) bf[fn] = ldB(sB, 32, wn + fn * 16, 0, lane);
#pragma unroll
    for (int fm = 0; fm < 4; ++fm) {
      v16h a = ldA(sA, 32, wm + fm * 16, 0, lane);
#pragma unroll
      for (int fn = 0; fn < 4; ++fn)
        acc[fm][fn] = wmma_f16(a, bf[fn], acc[fm][fn]);
    }
    __syncthreads();
    if (more) {
      stage();
      __syncthreads();
    }
  }
#pragma unroll
  for (int fm = 0; fm < 4; ++fm)
#pragma unroll
    for (int fn = 0; fn < 4; ++fn) {
      int col = n0 + wn + fn * 16 + (lane & 15);
      int rw = m0 + wm + fm * 16 + 8 * (lane >> 4);
#pragma unroll
      for (int r = 0; r < 8; ++r)
        C[(size_t)(rw + r) * N + col] = acc[fm][fn][r];
    }
}

// ---------------------------------------------------------------- beta = sigmoid(hs@Wb)
__global__ __launch_bounds__(256) void beta_kernel(const float* __restrict__ hs,
                                                   const float* __restrict__ Wb,
                                                   float* __restrict__ beta) {
  const int row = blockIdx.x;
  const int lane = threadIdx.x & 31, wid = threadIdx.x >> 5;
  const float* hr = hs + (size_t)row * H_;
  float a0 = 0, a1 = 0, a2 = 0, a3 = 0;
  for (int k = threadIdx.x; k < H_; k += 256) {
    float h = hr[k];
    a0 += h * Wb[k * NH_ + 0];
    a1 += h * Wb[k * NH_ + 1];
    a2 += h * Wb[k * NH_ + 2];
    a3 += h * Wb[k * NH_ + 3];
  }
  a0 = wred(a0); a1 = wred(a1); a2 = wred(a2); a3 = wred(a3);
  __shared__ float red[8][4];
  if (lane == 0) { red[wid][0] = a0; red[wid][1] = a1; red[wid][2] = a2; red[wid][3] = a3; }
  __syncthreads();
  if (threadIdx.x < 4) {
    float s = 0.f;
    for (int w = 0; w < 8; ++w) s += red[w][threadIdx.x];
    beta[(size_t)row * NH_ + threadIdx.x] = sigm(s);
  }
}

// ---------------------------------------------------------------- depthwise conv K=4 + SiLU
__global__ void conv_silu_kernel(const float* __restrict__ x,
                                 const float* __restrict__ w,
                                 float* __restrict__ y) {
  int t = blockIdx.x * blockDim.x + threadIdx.x;
  int ch = t & 1023;
  int l = (t >> 10) & (L_ - 1);
  int b = t >> 22;
  const float* wc = w + ch * 4;
  float s = 0.f;
#pragma unroll
  for (int j = 0; j < 4; ++j) {
    int ll = l - 3 + j;
    if (ll >= 0) s += x[((size_t)b * L_ + ll) * 1024 + ch] * wc[j];
  }
  y[t] = s * sigm(s);
}

// ---------------------------------------------------------------- depthwise FIR (no act)
__global__ void fir_kernel(const float* __restrict__ x,
                           const float* __restrict__ f, float* __restrict__ y,
                           int Kf) {
  int t = blockIdx.x * blockDim.x + threadIdx.x;
  int ch = t & 1023;
  int l = (t >> 10) & (L_ - 1);
  int b = t >> 22;
  const float* wc = f + ch * Kf;
  float s = 0.f;
  for (int j = 0; j < Kf; ++j) {
    int ll = l - (Kf - 1) + j;
    if (ll >= 0) s += x[((size_t)b * L_ + ll) * 1024 + ch] * wc[j];
  }
  y[t] = s;
}

// ---------------------------------------------------------------- delta rule chunkwise
// One WG per (b, h, dv-slice of 64). S slice (64x256 f32, dv-major) lives in
// LDS across all 128 sequential chunks. 8 waves / WG, ~152 KB LDS (<=160 KB so
// two WGs per 320KB WGP). All WMMA operands are b128 LDS loads.
__global__ __launch_bounds__(256) void delta_kernel(
    const float* __restrict__ qg, const float* __restrict__ kg,
    const float* __restrict__ vg, const float* __restrict__ betag,
    float* __restrict__ og) {
  __shared__ alignas(16) float    sS[DVB * DK_];   // 64 KB  S[dv][dk]  N-major
  __shared__ alignas(16) _Float16 sq[CH_ * DK_];   // 16 KB  q[ck][dk]  M-major
  __shared__ alignas(16) _Float16 sk[CH_ * DK_];   // 16 KB  k[ck][dk]  M-major / N-major for k^T
  __shared__ alignas(16) _Float16 skT[DK_ * CH_];  // 16 KB  k[dk][ck]  M-major for k^T, N-major for k_beta
  __shared__ alignas(16) _Float16 sw[CH_ * DK_];   // 16 KB  w[ck][dk]  M-major
  __shared__ alignas(16) _Float16 svt[DVB * CH_];  //  4 KB  v*beta [dv][ck] N-major
  __shared__ alignas(16) float    su[DVB * CH_];   //  8 KB  u [dv][ck] N-major
  __shared__ alignas(16) float    sT[CH_ * CH_];   //  4 KB  A then T (f32)
  __shared__ alignas(16) _Float16 sT16[CH_ * CH_]; //  2 KB  T as f16, M-major
  __shared__ alignas(16) _Float16 sat[CH_ * CH_];  //  2 KB  attn [ck][ck] M-major
  __shared__ alignas(16) _Float16 sua[DVB * CH_];  //  4 KB  u_adj [dv][ck] N-major
  __shared__ float sbeta[CH_];

  const int lane = threadIdx.x & 31, wid = threadIdx.x >> 5;
  const int dvb = blockIdx.x, h = blockIdx.y, b = blockIdx.z;
  const int dv0 = dvb * DVB;

  for (int i = threadIdx.x; i < DVB * DK_; i += 256) sS[i] = 0.f;
  __syncthreads();

  for (int c = 0; c < NCH_; ++c) {
    const int l0 = c * CH_;
    // ---- load + l2norm q/k, beta, v*beta slice; build skT, svt ----
    for (int rr = 0; rr < 4; ++rr) {
      int r = wid * 4 + rr;
      size_t roff = ((size_t)(b * L_ + l0 + r)) * (NH_ * DK_) + h * DK_;
      float bv = betag[(size_t)(b * L_ + l0 + r) * NH_ + h];
      if (lane == 0) sbeta[r] = bv;
      float tv[8], ss = 0.f;
#pragma unroll
      for (int i = 0; i < 8; ++i) { tv[i] = qg[roff + lane + 32 * i]; ss += tv[i] * tv[i]; }
      ss = wred(ss);
      float rn = rsqrtf(ss + 1e-6f);
#pragma unroll
      for (int i = 0; i < 8; ++i) sq[r * DK_ + lane + 32 * i] = (_Float16)(tv[i] * rn);
      ss = 0.f;
#pragma unroll
      for (int i = 0; i < 8; ++i) { tv[i] = kg[roff + lane + 32 * i]; ss += tv[i] * tv[i]; }
      ss = wred(ss);
      rn = rsqrtf(ss + 1e-6f);
#pragma unroll
      for (int i = 0; i < 8; ++i) {
        _Float16 hv = (_Float16)(tv[i] * rn);
        sk[r * DK_ + lane + 32 * i] = hv;
        skT[(lane + 32 * i) * CH_ + r] = hv;
      }
      size_t voff = ((size_t)(b * L_ + l0 + r)) * (NH_ * DV_) + h * DV_ + dv0;
#pragma unroll
      for (int i = 0; i < 2; ++i)
        svt[(lane + 32 * i) * CH_ + r] = (_Float16)(vg[voff + lane + 32 * i] * bv);
      if (c + 1 < NCH_) {
        __builtin_prefetch(qg + roff + (size_t)CH_ * NH_ * DK_, 0, 3);
        __builtin_prefetch(kg + roff + (size_t)CH_ * NH_ * DK_, 0, 3);
        __builtin_prefetch(vg + voff + (size_t)CH_ * NH_ * DV_, 0, 3);
      }
    }
    __syncthreads();

    // ---- A = -(k_beta @ k^T), strict lower (4 tiles, waves 0..3) ----
    if (wid < 4) {
      int m0 = (wid >> 1) * 16, n0 = (wid & 1) * 16;
      v8f acc = {};
      for (int kk = 0; kk < DK_; kk += 32)
        acc = wmma_f16(ldA_rs(sk, DK_, m0, kk, sbeta, lane),
                       ldB(sk, DK_, n0, kk, lane), acc);
      int col = n0 + (lane & 15), rb = m0 + 8 * (lane >> 4);
#pragma unroll
      for (int r = 0; r < 8; ++r) {
        int row = rb + r;
        sT[row * CH_ + col] = (row > col) ? -acc[r] : 0.f;
      }
    }
    __syncthreads();

    // ---- forward substitution -> T (wave 0; lane = column) ----
    if (wid == 0) {
      for (int i = 1; i < CH_; ++i) {
        float s = 0.f;
        for (int jj = 0; jj < i; ++jj) s += sT[i * CH_ + jj] * sT[jj * CH_ + lane];
        sT[i * CH_ + lane] += s;
      }
      sT[lane * CH_ + lane] += 1.f;
    }
    __syncthreads();
    for (int i = threadIdx.x; i < CH_ * CH_; i += 256) sT16[i] = (_Float16)sT[i];
    __syncthreads();

    // ---- u = T@v (1 tile/wave) ; w = T@k_beta (4 tiles/wave) ----
    {
      int m0 = (wid >> 2) * 16, n0 = (wid & 3) * 16;
      v8f z = {};
      v8f accu = wmma_f16(ldA(sT16, CH_, m0, 0, lane),
                          ldB(svt, CH_, n0, 0, lane), z);
      stC_nm(su, CH_, m0, n0, lane, accu);
#pragma unroll
      for (int tm = 0; tm < 2; ++tm) {
        v16h aT = ldA(sT16, CH_, tm * 16, 0, lane);
#pragma unroll
        for (int tt = 0; tt < 2; ++tt) {
          int n0w = (wid * 2 + tt) * 16;
          v8f zz = {};
          v8f accw = wmma_f16(aT, ldB_ks(skT, CH_, n0w, 0, sbeta, lane), zz);
          int colw = n0w + (lane & 15), rbw = tm * 16 + 8 * (lane >> 4);
#pragma unroll
          for (int r = 0; r < 8; ++r)
            sw[(rbw + r) * DK_ + colw] = (_Float16)accw[r];
        }
      }
    }
    __syncthreads();

    // ---- attn (waves 4..7) in parallel with u_adj = u - w@S (waves 0..3) ----
    if (wid >= 4) {
      int t = wid - 4;
      int m0 = (t >> 1) * 16, n0 = (t & 1) * 16;
      v8f acc = {};
      for (int kk = 0; kk < DK_; kk += 32)
        acc = wmma_f16(ldA(sq, DK_, m0, kk, lane),
                       ldB(sk, DK_, n0, kk, lane), acc);
      int col = n0 + (lane & 15), rb = m0 + 8 * (lane >> 4);
#pragma unroll
      for (int r = 0; r < 8; ++r) {
        int row = rb + r;
        sat[row * CH_ + col] = (_Float16)((row >= col) ? acc[r] : 0.f);
      }
    } else {
#pragma unroll
      for (int tm = 0; tm < 2; ++tm) {
        int m0 = tm * 16, n0 = wid * 16;
        v8f acc = {};
        for (int kk = 0; kk < DK_; kk += 32)
          acc = wmma_f16(ldA(sw, DK_, m0, kk, lane),
                         ldB_f32(sS, DK_, n0, kk, lane), acc);
        int col = n0 + (lane & 15), rb = m0 + 8 * (lane >> 4);
        const float* uq = su + col * CH_ + rb;
        v4f u0 = *(const v4f*)uq, u1 = *(const v4f*)(uq + 4);
        v8h o;
#pragma unroll
        for (int r = 0; r < 4; ++r) {
          o[r] = (_Float16)(u0[r] - acc[r]);
          o[4 + r] = (_Float16)(u1[r] - acc[4 + r]);
        }
        *(v8h*)(sua + col * CH_ + rb) = o;
      }
    }
    __syncthreads();

    // ---- o = q@S + attn@u_adj (1 tile/wave), store to global ----
    {
      int m0 = (wid >> 2) * 16, n0 = (wid & 3) * 16;
      v8f acc = {};
      for (int kk = 0; kk < DK_; kk += 32)
        acc = wmma_f16(ldA(sq, DK_, m0, kk, lane),
                       ldB_f32(sS, DK_, n0, kk, lane), acc);
      acc = wmma_f16(ldA(sat, CH_, m0, 0, lane),
                     ldB(sua, CH_, n0, 0, lane), acc);
      int col = n0 + (lane & 15), rb = m0 + 8 * (lane >> 4);
#pragma unroll
      for (int r = 0; r < 8; ++r) {
        int row = rb + r;
        og[((size_t)(b * L_ + l0 + row)) * (NH_ * DV_) + h * DV_ + dv0 + col] =
            acc[r];
      }
    }
    __syncthreads();

    // ---- S += k^T @ u_adj (64 tiles, 8/wave) ----
#pragma unroll
    for (int t = 0; t < 8; ++t) {
      int tile = wid * 8 + t;
      int m0 = (tile >> 2) * 16;  // dk
      int n0 = (tile & 3) * 16;   // dv
      v8f cfr = ldC_nm(sS, DK_, m0, n0, lane);
      v8f acc = wmma_f16(ldA(skT, CH_, m0, 0, lane),
                         ldB(sua, CH_, n0, 0, lane), cfr);
      stC_nm(sS, DK_, m0, n0, lane, acc);
    }
    __syncthreads();
  }
}

// ---------------------------------------------------------------- stats (mean,var,|x|,l2)
__global__ __launch_bounds__(256) void stats_kernel(
    const float* __restrict__ ls, const float* __restrict__ ll,
    const float* __restrict__ dd, const float* __restrict__ vv,
    float* __restrict__ stats) {
  const int lane = threadIdx.x & 31, wid = threadIdx.x >> 5;
  const int item = blockIdx.x * 8 + wid;  // (b*L)*NH + h
  const size_t base = (size_t)item * DV_;
  const float* ptrs[4] = {ls, ll, dd, vv};
  for (int tsr = 0; tsr < 4; ++tsr) {
    const float* p = ptrs[tsr] + base;
    float s1 = 0, s2 = 0, sa = 0;
#pragma unroll
    for (int i = 0; i < 8; ++i) {
      float x = p[lane + 32 * i];
      s1 += x; s2 += x * x; sa += fabsf(x);
    }
    s1 = wred(s1); s2 = wred(s2); sa = wred(sa);
    if (lane == 0) {
      float mean = s1 * (1.f / 256.f);
      float* o = stats + (size_t)item * 16 + tsr * 4;
      o[0] = mean;
      o[1] = s2 * (1.f / 256.f) - mean * mean;
      o[2] = sa * (1.f / 256.f);
      o[3] = sqrtf(s2);
    }
  }
}

// ---------------------------------------------------------------- gate MLP tail
__global__ __launch_bounds__(256) void gate_kernel(
    const float* __restrict__ G0, const float* __restrict__ stats,
    const float* __restrict__ gW1, const float* __restrict__ gb1,
    const float* __restrict__ gW2, const float* __restrict__ gb2,
    const float* __restrict__ log_temp, const float* __restrict__ floor_param,
    float* __restrict__ probs) {
  const int row = blockIdx.x;  // b*L
  const int lane = threadIdx.x & 31, wid = threadIdx.x >> 5;
  __shared__ float red[8][4];
  for (int h = 0; h < NH_; ++h) {
    float st[16];
    const float* sp = stats + ((size_t)row * NH_ + h) * 16;
#pragma unroll
    for (int i = 0; i < 16; ++i) st[i] = sp[i];
    float la[4] = {0, 0, 0, 0};
    for (int j = threadIdx.x; j < 1024; j += 256) {
      float hv = G0[(size_t)row * 1024 + j] + gb1[j];
#pragma unroll
      for (int s = 0; s < 16; ++s) hv += st[s] * gW1[(size_t)(1024 + s) * 1024 + j];
      float t = 0.5f * hv * (1.f + erff(hv * 0.70710678118f));
      la[0] += t * gW2[j * 4 + 0];
      la[1] += t * gW2[j * 4 + 1];
      la[2] += t * gW2[j * 4 + 2];
      la[3] += t * gW2[j * 4 + 3];
    }
#pragma unroll
    for (int cidx = 0; cidx < 4; ++cidx) la[cidx] = wred(la[cidx]);
    if (lane == 0) { red[wid][0] = la[0]; red[wid][1] = la[1]; red[wid][2] = la[2]; red[wid][3] = la[3]; }
    __syncthreads();
    if (threadIdx.x == 0) {
      float lg[4];
      for (int cidx = 0; cidx < 4; ++cidx) {
        float s = gb2[cidx];
        for (int w = 0; w < 8; ++w) s += red[w][cidx];
        lg[cidx] = s / expf(log_temp[h * 4 + cidx]);
      }
      float mx = fmaxf(fmaxf(lg[0], lg[1]), fmaxf(lg[2], lg[3]));
      float es[4], tot = 0.f;
      for (int cidx = 0; cidx < 4; ++cidx) { es[cidx] = expf(lg[cidx] - mx); tot += es[cidx]; }
      float fl[4], fsum = 0.f;
      for (int cidx = 0; cidx < 4; ++cidx) { fl[cidx] = 0.05f * sigm(floor_param[h * 4 + cidx]); fsum += fl[cidx]; }
      for (int cidx = 0; cidx < 4; ++cidx)
        probs[((size_t)row * NH_ + h) * 4 + cidx] =
            (es[cidx] / tot) * (1.f - fsum) + fl[cidx];
    }
    __syncthreads();
  }
}

// ---------------------------------------------------------------- combine + RMSNorm
__global__ __launch_bounds__(256) void combine_kernel(
    const float* __restrict__ ls, const float* __restrict__ ll,
    const float* __restrict__ dd, const float* __restrict__ vv,
    const float* __restrict__ idv, const float* __restrict__ probs,
    const float* __restrict__ alpha_id, const float* __restrict__ conv_res_logit,
    const float* __restrict__ o_norm_w, float* __restrict__ obuf) {
  const int lane = threadIdx.x & 31, wid = threadIdx.x >> 5;
  const int item = blockIdx.x * 8 + wid;  // (b*L)*NH + h
  const int h = item & (NH_ - 1);
  const size_t base = (size_t)item * DV_;
  const float* p = probs + (size_t)item * 4;
  float p0 = p[0], p1 = p[1], p2 = p[2], p3 = p[3];
  float cres = sigm(conv_res_logit[h]);
  float aid = alpha_id[h];
  float ov[8], ss = 0.f;
#pragma unroll
  for (int i = 0; i < 8; ++i) {
    size_t idx = base + lane + 32 * i;
    float o = p0 * ls[idx] + p1 * ll[idx] + p2 * dd[idx] + p3 * vv[idx] +
              cres * ls[idx] + aid * idv[idx];
    ov[i] = o;
    ss += o * o;
  }
  ss = wred(ss);
  float sc = rsqrtf(ss * (1.f / 256.f) + 1e-5f);
#pragma unroll
  for (int i = 0; i < 8; ++i) {
    int d = lane + 32 * i;
    obuf[base + d] = ov[i] * sc * o_norm_w[d];
  }
}

// ---------------------------------------------------------------- launch
extern "C" void kernel_launch(void* const* d_in, const int* in_sizes, int n_in,
                              void* d_out, int out_size, void* d_ws,
                              size_t ws_size, hipStream_t stream) {
  (void)in_sizes; (void)n_in; (void)out_size; (void)ws_size;
  const float* hs   = (const float*)d_in[0];
  const float* Wq   = (const float*)d_in[1];
  const float* Wk   = (const float*)d_in[2];
  const float* Wv   = (const float*)d_in[3];
  const float* cwq  = (const float*)d_in[4];
  const float* cwk  = (const float*)d_in[5];
  const float* cwv  = (const float*)d_in[6];
  const float* Wb   = (const float*)d_in[7];
  const float* Wid  = (const float*)d_in[8];
  const float* alpha_id = (const float*)d_in[9];
  const float* firs = (const float*)d_in[10];
  const float* firl = (const float*)d_in[11];
  const float* gW1  = (const float*)d_in[12];
  const float* gb1  = (const float*)d_in[13];
  const float* gW2  = (const float*)d_in[14];
  const float* gb2  = (const float*)d_in[15];
  const float* log_temp = (const float*)d_in[16];
  const float* floor_param = (const float*)d_in[17];
  const float* conv_res_logit = (const float*)d_in[18];
  const float* o_norm_w = (const float*)d_in[19];
  const float* Wo   = (const float*)d_in[20];
  float* out = (float*)d_out;
  float* ws = (float*)d_ws;

  const size_t NBLC = (size_t)B_ * L_ * 1024;
  size_t off = 0;
  float* qp   = ws + off; off += NBLC;
  float* kp   = ws + off; off += NBLC;
  float* vp   = ws + off; off += NBLC;
  float* q    = ws + off; off += NBLC;
  float* k    = ws + off; off += NBLC;
  float* v    = ws + off; off += NBLC;
  float* idv  = ws + off; off += NBLC;
  float* dout = ws + off; off += NBLC;
  float* fs   = ws + off; off += NBLC;
  float* fl   = ws + off; off += NBLC;
  float* G0   = ws + off; off += NBLC;
  float* obuf = ws + off; off += NBLC;
  float* beta  = ws + off; off += (size_t)B_ * L_ * NH_;
  float* stats = ws + off; off += (size_t)B_ * L_ * NH_ * 16;
  float* probs = ws + off; off += (size_t)B_ * L_ * NH_ * 4;

  // f16 pre-transposed weights (N-major [N][K]) carved after the f32 buffers
  const size_t WSZ = (size_t)1024 * 1024;
  _Float16* wsh = (_Float16*)(ws + off);
  _Float16* WqT  = wsh + 0 * WSZ;
  _Float16* WkT  = wsh + 1 * WSZ;
  _Float16* WvT  = wsh + 2 * WSZ;
  _Float16* WidT = wsh + 3 * WSZ;
  _Float16* gW1T = wsh + 4 * WSZ;
  _Float16* WoT  = wsh + 5 * WSZ;

  const int M = B_ * L_;
  dim3 gTr(1024 / 32, 1024 / 32);
  wtrans_kernel<<<gTr, 256, 0, stream>>>(Wq, WqT, 1024, 1024);
  wtrans_kernel<<<gTr, 256, 0, stream>>>(Wk, WkT, 1024, 1024);
  wtrans_kernel<<<gTr, 256, 0, stream>>>(Wv, WvT, 1024, 1024);
  wtrans_kernel<<<gTr, 256, 0, stream>>>(Wid, WidT, 1024, 1024);
  wtrans_kernel<<<gTr, 256, 0, stream>>>(gW1, gW1T, 1024, 1024);  // top 1024 rows
  wtrans_kernel<<<gTr, 256, 0, stream>>>(Wo, WoT, 1024, 1024);

  dim3 gGemm(1024 / 256, M / 128);
  gemm_wmma_kernel<<<gGemm, 256, 0, stream>>>(hs, WqT, qp, M, 1024, 1024);
  gemm_wmma_kernel<<<gGemm, 256, 0, stream>>>(hs, WkT, kp, M, 1024, 1024);
  gemm_wmma_kernel<<<gGemm, 256, 0, stream>>>(hs, WvT, vp, M, 1024, 1024);
  gemm_wmma_kernel<<<gGemm, 256, 0, stream>>>(hs, WidT, idv, M, 1024, 1024);
  gemm_wmma_kernel<<<gGemm, 256, 0, stream>>>(hs, gW1T, G0, M, 1024, 1024);

  beta_kernel<<<M, 256, 0, stream>>>(hs, Wb, beta);

  const int totc = B_ * L_ * 1024;
  conv_silu_kernel<<<totc / 256, 256, 0, stream>>>(qp, cwq, q);
  conv_silu_kernel<<<totc / 256, 256, 0, stream>>>(kp, cwk, k);
  conv_silu_kernel<<<totc / 256, 256, 0, stream>>>(vp, cwv, v);

  delta_kernel<<<dim3(NDVB, NH_, B_), 256, 0, stream>>>(q, k, v, beta, dout);

  fir_kernel<<<totc / 256, 256, 0, stream>>>(v, firs, fs, 3);
  fir_kernel<<<totc / 256, 256, 0, stream>>>(v, firl, fl, 63);

  stats_kernel<<<(M * NH_) / 8, 256, 0, stream>>>(fs, fl, dout, v, stats);
  gate_kernel<<<M, 256, 0, stream>>>(G0, stats, gW1, gb1, gW2, gb2, log_temp,
                                     floor_param, probs);
  combine_kernel<<<(M * NH_) / 8, 256, 0, stream>>>(
      fs, fl, dout, v, idv, probs, alpha_id, conv_res_logit, o_norm_w, obuf);

  gemm_wmma_kernel<<<gGemm, 256, 0, stream>>>(obuf, WoT, out, M, 1024, 1024);
}
